// DeformableTransformerEncoderLayer_68169720922404
// MI455X (gfx1250) — compile-verified, matrix operand order checked by reading, not run on hardware
//
#include <hip/hip_runtime.h>
#include <hip/hip_bf16.h>
#include <stddef.h>

// ---------------------------------------------------------------------------
// DeformableTransformerEncoderLayer for MI455X (gfx1250, wave32, WMMA + TDM).
// Dense GEMMs: v_wmma_f32_16x16x32_bf16; LDS tiles fed by the Tensor Data
// Mover (tensor_load_to_lds, TENSORcnt), triple-buffered with a depth-2
// DMA pipeline (s_wait_tensorcnt 2), fused bias/relu/residual/bf16 epilogues.
// Deformable attention is a wave32 gather kernel. Pipeline is HBM-bandwidth
// bound; every intermediate touches HBM exactly once.
// ---------------------------------------------------------------------------

typedef unsigned short u16;
typedef __bf16 bf16_t;
typedef bf16_t v16bf __attribute__((ext_vector_type(16)));
typedef float  v8f   __attribute__((ext_vector_type(8)));
typedef unsigned int v4u __attribute__((ext_vector_type(4)));
typedef int v8i __attribute__((ext_vector_type(8)));
typedef int v4i __attribute__((ext_vector_type(4)));

static constexpr int cB  = 16;
static constexpr int cM  = 2048;
static constexpr int cD  = 512;
static constexpr int cH  = 8;
static constexpr int cK  = 5;
static constexpr int cHD = 64;
static constexpr int cN  = cB * cM;          // 32768 rows

#if defined(__gfx1250__) && __has_builtin(__builtin_amdgcn_tensor_load_to_lds)
#define HAVE_TDM 1
#else
#define HAVE_TDM 0
#endif

__device__ __forceinline__ u16 f32_to_bf16(float f) {
  union { float f; unsigned u; } v; v.f = f;
  unsigned r = v.u + 0x7FFFu + ((v.u >> 16) & 1u);   // RNE
  return (u16)(r >> 16);
}

// Build a v16bf A/B fragment from LDS: 8 contiguous bf16 at [kb] and [kb+16].
__device__ __forceinline__ v16bf frag_ld(const u16* row, int kb) {
  union { uint4 q[2]; v16bf v; } u;
  u.q[0] = *(const uint4*)(row + kb);
  u.q[1] = *(const uint4*)(row + kb + 16);
  return u.v;
}

__device__ __forceinline__ v8f wmma_bf16(v16bf a, v16bf b, v8f c) {
  return __builtin_amdgcn_wmma_f32_16x16x32_bf16(
      /*neg_a=*/false, a, /*neg_b=*/false, b,
      /*c_mod=*/(short)0, c, /*reuse_a=*/false, /*reuse_b=*/false);
}

#if HAVE_TDM
// 32-bit LDS byte offset of a shared-memory pointer (addrspace(3) cast).
typedef __attribute__((address_space(3))) const void lds_cv;
__device__ __forceinline__ unsigned lds_addr32(const void* p) {
  return (unsigned)(size_t)(lds_cv*)p;
}

// TDM DMA of a 2D bf16 tile: 128 rows x 32 elems, row stride K elems, into
// LDS rows padded 16 DWORDs data + 4 DWORDs pad (pad_interval=3, pad_amount=3)
// to match the [128][40] u16 LDS layout used by the WMMA fragment loads.
__device__ __forceinline__ void tdm_tile(const u16* gsrc, unsigned lds_off, int K) {
  unsigned long long ga = (unsigned long long)(size_t)gsrc;
  v4u g0 = { 1u,                                       // count=1 (valid D#)
             lds_off,                                  // lds_addr
             (unsigned)ga,                             // global_addr[31:0]
             (unsigned)((ga >> 32) & 0x1FFFFFFu) | (2u << 30) };  // ga[56:32] | type=2
  v8i g1 = { (int)((1u << 16) | (1u << 20) | (3u << 22) | (3u << 25)),
             //  data_size=2B | pad_enable | pad_interval=16DW | pad_amount=4DW
             (int)((unsigned)K << 16),                 // tensor_dim0[15:0]
             (int)(0x40000000u | ((unsigned)K >> 16)), // tensor_dim0 hi | tensor_dim1=16384
             (int)(32u << 16),                         // tensor_dim1 hi=0 | tile_dim0=32
             128,                                      // tile_dim1=128, tile_dim2=0
             (int)K,                                   // tensor_dim0_stride[31:0]
             0, 0 };                                   // stride hi, dim1_stride
  v4i g2 = { 0, 0, 0, 0 };
  v4i g3 = { 0, 0, 0, 0 };
#if __clang_major__ >= 23
  v8i g4 = { 0, 0, 0, 0, 0, 0, 0, 0 };
  __builtin_amdgcn_tensor_load_to_lds(g0, g1, g2, g3, g4, 0);
#else
  __builtin_amdgcn_tensor_load_to_lds(g0, g1, g2, g3, 0);
#endif
}
#endif  // HAVE_TDM

// ---------------------------------------------------------------------------
// Generic WMMA GEMM:  out[N x Dout] = act( A_bf16[N x K] @ BT_bf16[Dout x K]^T
//                                          + bias ) + resid
// Tile 128x128, 256 threads (8 waves: 4 over M, 2 over N), TK=32. Triple-
// buffered LDS, TDM pipeline depth 2: wait tensorcnt<=2 keeps one tile in
// flight while computing the current one.
// ---------------------------------------------------------------------------
__global__ __launch_bounds__(256) void gemm_bf16_wmma(
    const u16* __restrict__ Aa, const u16* __restrict__ Bt,
    const float* __restrict__ bias, const float* __restrict__ resid,
    float* __restrict__ outF, u16* __restrict__ outBF,
    int N, int K, int Dout, int relu)
{
  __shared__ __align__(16) u16 lA[3][128][40];   // 32 K + 8 pad (80B rows)
  __shared__ __align__(16) u16 lB[3][128][40];

  const int m0 = blockIdx.x * 128;
  const int n0 = blockIdx.y * 128;
  const int tid  = threadIdx.x;
  const int wave = tid >> 5;
  const int lane = tid & 31;
  const int mbase = (wave & 3) * 32;   // 4 waves along M
  const int nbase = (wave >> 2) * 64;  // 2 waves along N

  v8f acc[2][4] = {};
  const int ksteps = K >> 5;

#if HAVE_TDM
  if (tid == 0) {                       // prime pipeline: tiles 0 and 1
    tdm_tile(Aa + (size_t)m0 * K, lds_addr32(&lA[0][0][0]), K);
    tdm_tile(Bt + (size_t)n0 * K, lds_addr32(&lB[0][0][0]), K);
    if (ksteps > 1) {
      tdm_tile(Aa + (size_t)m0 * K + 32, lds_addr32(&lA[1][0][0]), K);
      tdm_tile(Bt + (size_t)n0 * K + 32, lds_addr32(&lB[1][0][0]), K);
    }
  }
#else
  auto loadTile = [&](int buf, int kk) {
#pragma unroll
    for (int j = 0; j < 2; ++j) {
      int i = tid + 256 * j;           // 0..511 -> 128 rows x 4 chunks
      int r = i >> 2;
      int c = (i & 3) * 8;
      *(uint4*)&lA[buf][r][c] = *(const uint4*)(Aa + (size_t)(m0 + r) * K + kk + c);
      *(uint4*)&lB[buf][r][c] = *(const uint4*)(Bt + (size_t)(n0 + r) * K + kk + c);
    }
  };
  loadTile(0, 0);
#endif

  for (int kt = 0; kt < ksteps; ++kt) {
#if HAVE_TDM
    if (kt + 1 < ksteps) __builtin_amdgcn_s_wait_tensorcnt(2);  // tile kt done,
    else                 __builtin_amdgcn_s_wait_tensorcnt(0);  // kt+1 in flight
    __syncthreads();                     // publish LDS to all 8 waves
    if (kt + 2 < ksteps && tid == 0) {   // DMA tile kt+2 under compute
      const int kk = (kt + 2) << 5;
      const int b  = (kt + 2) % 3;
      tdm_tile(Aa + (size_t)m0 * K + kk, lds_addr32(&lA[b][0][0]), K);
      tdm_tile(Bt + (size_t)n0 * K + kk, lds_addr32(&lB[b][0][0]), K);
    }
#else
    __syncthreads();
    if (kt + 1 < ksteps) loadTile((kt + 1) % 3, (kt + 1) << 5);
#endif

    const int buf = kt % 3;
    const int kb  = (lane >> 4) * 8;      // K halves split at lane 16
    const int rA  = mbase + (lane & 15);
    const int rB  = nbase + (lane & 15);

    v16bf a0 = frag_ld(&lA[buf][rA][0],      kb);
    v16bf a1 = frag_ld(&lA[buf][rA + 16][0], kb);
    v16bf b0 = frag_ld(&lB[buf][rB][0],      kb);
    v16bf b1 = frag_ld(&lB[buf][rB + 16][0], kb);
    v16bf b2 = frag_ld(&lB[buf][rB + 32][0], kb);
    v16bf b3 = frag_ld(&lB[buf][rB + 48][0], kb);

    acc[0][0] = wmma_bf16(a0, b0, acc[0][0]);
    acc[0][1] = wmma_bf16(a0, b1, acc[0][1]);
    acc[0][2] = wmma_bf16(a0, b2, acc[0][2]);
    acc[0][3] = wmma_bf16(a0, b3, acc[0][3]);
    acc[1][0] = wmma_bf16(a1, b0, acc[1][0]);
    acc[1][1] = wmma_bf16(a1, b1, acc[1][1]);
    acc[1][2] = wmma_bf16(a1, b2, acc[1][2]);
    acc[1][3] = wmma_bf16(a1, b3, acc[1][3]);
  }

  // Epilogue: C/D layout -> lanes 0-15: M=i, N=lane; lanes 16-31: M=8+i.
#pragma unroll
  for (int ms = 0; ms < 2; ++ms) {
#pragma unroll
    for (int ns = 0; ns < 4; ++ns) {
      const int col = n0 + nbase + ns * 16 + (lane & 15);
      const float bvv = bias ? bias[col] : 0.0f;
#pragma unroll
      for (int i = 0; i < 8; ++i) {
        const int row = m0 + mbase + ms * 16 + (lane >> 4) * 8 + i;
        float v = acc[ms][ns][i] + bvv;
        if (relu) v = fmaxf(v, 0.0f);
        const size_t idx = (size_t)row * Dout + col;
        if (resid) v += resid[idx];
        if (outF) outF[idx] = v;
        else      outBF[idx] = f32_to_bf16(v);
      }
    }
  }
}

// ---------------------------------------------------------------------------
// LayerNorm: one wave per 512-wide row, lane holds 16 elems, shfl-reduce.
// ---------------------------------------------------------------------------
__global__ __launch_bounds__(256) void layernorm_kernel(
    const float* __restrict__ x, const float* __restrict__ g,
    const float* __restrict__ bta, u16* __restrict__ out)
{
  const int wave = threadIdx.x >> 5, lane = threadIdx.x & 31;
  const size_t row = (size_t)blockIdx.x * 8 + wave;
  const float* xr = x + row * cD + lane * 16;
  float v[16];
#pragma unroll
  for (int j = 0; j < 4; ++j) {
    float4 f = *(const float4*)(xr + j * 4);
    v[j*4+0] = f.x; v[j*4+1] = f.y; v[j*4+2] = f.z; v[j*4+3] = f.w;
  }
  float s = 0.f, s2 = 0.f;
#pragma unroll
  for (int i = 0; i < 16; ++i) { s += v[i]; s2 += v[i] * v[i]; }
#pragma unroll
  for (int m = 16; m > 0; m >>= 1) { s += __shfl_xor(s, m, 32); s2 += __shfl_xor(s2, m, 32); }
  const float mean = s * (1.0f / cD);
  const float var  = s2 * (1.0f / cD) - mean * mean;
  const float inv  = rsqrtf(var + 1e-6f);
  union { u16 u[16]; uint4 q[2]; } o;
#pragma unroll
  for (int i = 0; i < 16; ++i) {
    const int c = lane * 16 + i;
    o.u[i] = f32_to_bf16((v[i] - mean) * inv * g[c] + bta[c]);
  }
  uint4* dst = (uint4*)(out + row * cD + lane * 16);
  dst[0] = o.q[0]; dst[1] = o.q[1];
}

// ---------------------------------------------------------------------------
// AvgPool1d(kernel=5, stride=1, pad=2, count_include_pad) over seq dim.
// ---------------------------------------------------------------------------
__global__ __launch_bounds__(256) void pool_kernel(
    const float* __restrict__ qraw, float* __restrict__ qbar)
{
  const size_t idx = (size_t)blockIdx.x * 256 + threadIdx.x;   // < N*D
  const int d = (int)(idx & (cD - 1));
  const size_t row = idx >> 9;
  const int m = (int)(row & (cM - 1));
  const size_t brow = row - m;
  float s = 0.f;
#pragma unroll
  for (int w = -2; w <= 2; ++w) {
    const int mm = m + w;
    if (mm >= 0 && mm < cM) s += qraw[(brow + mm) * cD + d];
  }
  qbar[idx] = s * 0.2f;
}

// sign_lens[b] = sum(mask[b]) - 1
__global__ __launch_bounds__(256) void lens_kernel(
    const unsigned char* __restrict__ mask, float* __restrict__ lens)
{
  __shared__ int sb[256];
  const int b = blockIdx.x, t = threadIdx.x;
  int s = 0;
  for (int i = t; i < cM; i += 256) s += mask[(size_t)b * cM + i] ? 1 : 0;
  sb[t] = s; __syncthreads();
  for (int st = 128; st > 0; st >>= 1) { if (t < st) sb[t] += sb[t + st]; __syncthreads(); }
  if (t == 0) lens[b] = (float)sb[0] - 1.0f;
}

// offsets[row, c] = qbar[row,:] . WoffT[c,:] + boff[c]  (wave per row, 40 cols)
__global__ __launch_bounds__(256) void offsets_kernel(
    const float* __restrict__ qbar, const float* __restrict__ wofft,
    const float* __restrict__ boff, float* __restrict__ offs)
{
  const int wave = threadIdx.x >> 5, lane = threadIdx.x & 31;
  const size_t r = (size_t)blockIdx.x * 8 + wave;
  const float* qrow = qbar + r * cD + lane * 16;
  float qv[16];
#pragma unroll
  for (int i = 0; i < 16; ++i) qv[i] = qrow[i];
  for (int c = 0; c < cH * cK; ++c) {
    const float* wr = wofft + (size_t)c * cD + lane * 16;
    float p = 0.f;
#pragma unroll
    for (int i = 0; i < 16; ++i) p += qv[i] * wr[i];
#pragma unroll
    for (int m = 16; m > 0; m >>= 1) p += __shfl_xor(p, m, 32);
    if (lane == 0) offs[r * (cH * cK) + c] = p + boff[c];
  }
}

// ---------------------------------------------------------------------------
// Deformable sampling + softmax(K=5) + context. One wave per (b,h,m);
// lane owns 2 of the 64 head channels.
// ---------------------------------------------------------------------------
__global__ __launch_bounds__(256) void attn_kernel(
    const float* __restrict__ kbuf, const float* __restrict__ vbuf,
    const float* __restrict__ qbar, const float* __restrict__ offs,
    const float* __restrict__ lens, u16* __restrict__ ctx)
{
  const int wave = threadIdx.x >> 5, lane = threadIdx.x & 31;
  const int gid = blockIdx.x * 8 + wave;            // b*H*M + h*M + m
  const int b = gid >> 14;                          // H*M = 16384
  const int h = (gid >> 11) & (cH - 1);
  const int m = gid & (cM - 1);
  const int c0 = lane * 2, c1 = c0 + 1;

  const size_t rowq = ((size_t)b * cM + m) * cD + h * cHD;
  const float scale = 0.125f;                       // 1/sqrt(64)
  const float q0 = qbar[rowq + c0] * scale;
  const float q1 = qbar[rowq + c1] * scale;
  const float len = lens[b];

  float scr[cK], w1a[cK];
  int   i0a[cK];

#pragma unroll
  for (int kk = 0; kk < cK; ++kk) {
    const float off = offs[((size_t)b * cM + m) * (cH * cK) + h * cK + kk];
    float sl = (float)(kk - 3) + off + (float)m;    // ref = [-3..1]
    sl = sl - floorf(sl / len) * len;               // python mod (len>0)
    const float ix = sl * ((float)cM / (float)(cM - 1)) - 0.5f;
    const float x0 = floorf(ix);
    const float w1 = ix - x0;
    const int i0 = (int)x0;
    w1a[kk] = w1; i0a[kk] = i0;

    const float* kr = kbuf + ((size_t)b * cM + i0) * cD + h * cHD;
    const bool v0 = (unsigned)i0 < (unsigned)cM;
    const bool v1 = (unsigned)(i0 + 1) < (unsigned)cM;
    const float k00 = v0 ? kr[c0] : 0.f;
    const float k01 = v0 ? kr[c1] : 0.f;
    const float k10 = v1 ? kr[cD + c0] : 0.f;
    const float k11 = v1 ? kr[cD + c1] : 0.f;
    const float sk0 = ((1.f - w1) * k00 + w1 * k10) * 0.5f;
    const float sk1 = ((1.f - w1) * k01 + w1 * k11) * 0.5f;

    float p = q0 * sk0 + q1 * sk1;
#pragma unroll
    for (int s = 16; s > 0; s >>= 1) p += __shfl_xor(p, s, 32);
    scr[kk] = p;
  }

  float mx = scr[0];
#pragma unroll
  for (int kk = 1; kk < cK; ++kk) mx = fmaxf(mx, scr[kk]);
  float e[cK], den = 0.f;
#pragma unroll
  for (int kk = 0; kk < cK; ++kk) { e[kk] = __expf(scr[kk] - mx); den += e[kk]; }
  const float rden = 1.0f / den;

  float a0 = 0.f, a1 = 0.f;
#pragma unroll
  for (int kk = 0; kk < cK; ++kk) {
    const float wgt = e[kk] * rden;
    const int i0 = i0a[kk];
    const float w1 = w1a[kk];
    const float* vr = vbuf + ((size_t)b * cM + i0) * cD + h * cHD;
    const bool v0 = (unsigned)i0 < (unsigned)cM;
    const bool v1 = (unsigned)(i0 + 1) < (unsigned)cM;
    const float p00 = v0 ? vr[c0] : 0.f;
    const float p01 = v0 ? vr[c1] : 0.f;
    const float p10 = v1 ? vr[cD + c0] : 0.f;
    const float p11 = v1 ? vr[cD + c1] : 0.f;
    a0 += wgt * ((1.f - w1) * p00 + w1 * p10) * 0.5f;
    a1 += wgt * ((1.f - w1) * p01 + w1 * p11) * 0.5f;
  }
  const unsigned packed = (unsigned)f32_to_bf16(a0) | ((unsigned)f32_to_bf16(a1) << 16);
  *(unsigned*)&ctx[rowq + c0] = packed;
}

// Weight converters: W[din x dout] (f32) -> WT[dout x din] (bf16 / f32)
__global__ __launch_bounds__(256) void convertT_kernel(
    const float* __restrict__ W, u16* __restrict__ WT, int din, int dout)
{
  const int idx = blockIdx.x * 256 + threadIdx.x;
  if (idx >= din * dout) return;
  const int k = idx / dout, c = idx - k * dout;
  WT[(size_t)c * din + k] = f32_to_bf16(W[idx]);
}

__global__ __launch_bounds__(256) void transposeF_kernel(
    const float* __restrict__ W, float* __restrict__ WT, int din, int dout)
{
  const int idx = blockIdx.x * 256 + threadIdx.x;
  if (idx >= din * dout) return;
  const int k = idx / dout, c = idx - k * dout;
  WT[(size_t)c * din + k] = W[idx];
}

// ---------------------------------------------------------------------------
// Launch orchestration. Workspace layout (~338 MB required):
//   [0,32M)    xn_bf16            -> reused as on_bf16 after attention
//   [32,96M)   k_f32              -> reused (with v) as ffn hidden bf16
//   [96,160M)  v_f32
//   [160,224M) qbar_f32
//   [224,232M) offsets f32
//   [232,264M) ctx_bf16
//   [264,328M) q_raw f32 -> o f32
//   [328M..)   bf16 weights, WoffT f32, lens
// ---------------------------------------------------------------------------
extern "C" void kernel_launch(void* const* d_in, const int* in_sizes, int n_in,
                              void* d_out, int out_size, void* d_ws, size_t ws_size,
                              hipStream_t stream) {
  (void)in_sizes; (void)n_in; (void)out_size; (void)ws_size;
  const float* x     = (const float*)d_in[0];
  const unsigned char* mask = (const unsigned char*)d_in[1];
  const float* ln1_g = (const float*)d_in[2];
  const float* ln1_b = (const float*)d_in[3];
  const float* Wk    = (const float*)d_in[4];
  const float* bk    = (const float*)d_in[5];
  const float* Wv    = (const float*)d_in[6];
  const float* bv    = (const float*)d_in[7];
  const float* Wq    = (const float*)d_in[8];
  const float* bq    = (const float*)d_in[9];
  const float* Woff  = (const float*)d_in[10];
  const float* boff  = (const float*)d_in[11];
  const float* Wo    = (const float*)d_in[12];
  const float* bo    = (const float*)d_in[13];
  const float* ln2_g = (const float*)d_in[14];
  const float* ln2_b = (const float*)d_in[15];
  const float* W1    = (const float*)d_in[16];
  const float* b1    = (const float*)d_in[17];
  const float* W2    = (const float*)d_in[18];
  const float* b2    = (const float*)d_in[19];
  float* out = (float*)d_out;

  char* ws = (char*)d_ws;
  auto MB = [](size_t m) { return m << 20; };
  u16*   xn    = (u16*)(ws + 0);
  float* kbuf  = (float*)(ws + MB(32));
  float* vbuf  = (float*)(ws + MB(96));
  float* qbar  = (float*)(ws + MB(160));
  float* offs  = (float*)(ws + MB(224));
  u16*   ctx   = (u16*)(ws + MB(232));
  float* obuf  = (float*)(ws + MB(264));   // q_raw, then o
  u16*   onb   = (u16*)(ws + 0);           // reuse xn region
  u16*   hff   = (u16*)(ws + MB(32));      // reuse k/v regions
  u16*   wkT   = (u16*)(ws + MB(328));
  u16*   wvT   = wkT + 512 * 512;
  u16*   wqT   = wvT + 512 * 512;
  u16*   woT   = wqT + 512 * 512;
  u16*   w1T   = woT + 512 * 512;          // 2048 x 512
  u16*   w2T   = w1T + 512 * 2048;         // 512 x 2048
  float* woffT = (float*)(ws + MB(336));
  float* lens  = (float*)(ws + MB(337));

  // Weight prep (bf16 transposed for contiguous B-operand K runs)
  convertT_kernel<<<(512 * 512) / 256, 256, 0, stream>>>(Wk, wkT, 512, 512);
  convertT_kernel<<<(512 * 512) / 256, 256, 0, stream>>>(Wv, wvT, 512, 512);
  convertT_kernel<<<(512 * 512) / 256, 256, 0, stream>>>(Wq, wqT, 512, 512);
  convertT_kernel<<<(512 * 512) / 256, 256, 0, stream>>>(Wo, woT, 512, 512);
  convertT_kernel<<<(512 * 2048) / 256, 256, 0, stream>>>(W1, w1T, 512, 2048);
  convertT_kernel<<<(512 * 2048) / 256, 256, 0, stream>>>(W2, w2T, 2048, 512);
  transposeF_kernel<<<(512 * 40 + 255) / 256, 256, 0, stream>>>(Woff, woffT, 512, 40);
  lens_kernel<<<cB, 256, 0, stream>>>(mask, lens);

  // LN1 -> xn (bf16)
  layernorm_kernel<<<cN / 8, 256, 0, stream>>>(x, ln1_g, ln1_b, xn);

  // k, v, q_raw GEMMs (WMMA bf16, TDM-fed)
  gemm_bf16_wmma<<<dim3(cN / 128, 4), 256, 0, stream>>>(xn, wkT, bk, nullptr, kbuf, nullptr, cN, 512, 512, 0);
  gemm_bf16_wmma<<<dim3(cN / 128, 4), 256, 0, stream>>>(xn, wvT, bv, nullptr, vbuf, nullptr, cN, 512, 512, 0);
  gemm_bf16_wmma<<<dim3(cN / 128, 4), 256, 0, stream>>>(xn, wqT, bq, nullptr, obuf, nullptr, cN, 512, 512, 0);

  // q moving-mean, offsets, deformable attention
  pool_kernel<<<(cN * cD) / 256, 256, 0, stream>>>(obuf, qbar);
  offsets_kernel<<<cN / 8, 256, 0, stream>>>(qbar, woffT, boff, offs);
  attn_kernel<<<(cB * cH * cM) / 8, 256, 0, stream>>>(kbuf, vbuf, qbar, offs, lens, ctx);

  // o = ctx @ Wo + bo + x  (fused residual)
  gemm_bf16_wmma<<<dim3(cN / 128, 4), 256, 0, stream>>>(ctx, woT, bo, x, obuf, nullptr, cN, 512, 512, 0);

  // LN2 -> on (bf16)
  layernorm_kernel<<<cN / 8, 256, 0, stream>>>(obuf, ln2_g, ln2_b, onb);

  // h = relu(on @ W1 + b1) (bf16), out = h @ W2 + b2 + o (f32)
  gemm_bf16_wmma<<<dim3(cN / 128, 16), 256, 0, stream>>>(onb, w1T, b1, nullptr, nullptr, hff, cN, 512, 2048, 1);
  gemm_bf16_wmma<<<dim3(cN / 128, 4), 256, 0, stream>>>(hff, w2T, b2, obuf, out, nullptr, cN, 2048, 512, 0);
}
